// SMLP_eg_67405216744006
// MI455X (gfx1250) — compile-verified, matrix operand order checked by reading, not run on hardware
//
#include <hip/hip_runtime.h>
#include <hip/hip_bf16.h>
#include <stdint.h>

typedef __attribute__((ext_vector_type(16))) _Float16 v16h;
typedef __attribute__((ext_vector_type(8)))  _Float16 v8h;
typedef __attribute__((ext_vector_type(4)))  _Float16 v4h;
typedef __attribute__((ext_vector_type(8)))  float    v8f;
typedef int v4i __attribute__((vector_size(16)));

#define HAS_ASYNC_LDS __has_builtin(__builtin_amdgcn_global_load_async_to_lds_b128)

// Stage one 16B chunk global -> LDS. Async (ASYNCcnt-tracked) when the
// builtin exists, else synchronous load + ds_store fallback.
// Signature (from clang diagnostic): (as1 v4i*, as3 v4i*, i32 offset, i32 cpol)
__device__ __forceinline__ void stage16(const _Float16* g, _Float16* l) {
#if HAS_ASYNC_LDS
  typedef __attribute__((address_space(1))) v4i* gv4;
  typedef __attribute__((address_space(3))) v4i* lv4;
  __builtin_amdgcn_global_load_async_to_lds_b128(
      (gv4)(uintptr_t)g, (lv4)(uintptr_t)l, /*offset=*/0, /*cpol=*/0);
#else
  *(v8h*)l = *(const v8h*)g;
#endif
}

__device__ __forceinline__ void wait_stage() {
#if HAS_ASYNC_LDS
#if __has_builtin(__builtin_amdgcn_s_wait_asynccnt)
  __builtin_amdgcn_s_wait_asynccnt(0);
#else
  asm volatile("s_wait_asynccnt 0x0" ::: "memory");
#endif
#endif
}

// ---------------------------------------------------------------------------
// f32 -> f16 conversion (vectorized x4)
// ---------------------------------------------------------------------------
__global__ void smlp_cvt_f16_kernel(const float* __restrict__ src,
                                    _Float16* __restrict__ dst, int n) {
  int i = (blockIdx.x * blockDim.x + threadIdx.x) * 4;
  if (i + 3 < n) {
    float4 f = *(const float4*)(src + i);
    v4h h;
    h[0] = (_Float16)f.x;
    h[1] = (_Float16)f.y;
    h[2] = (_Float16)f.z;
    h[3] = (_Float16)f.w;
    *(v4h*)(dst + i) = h;
  }
}

// ---------------------------------------------------------------------------
// LIF scan over T; spikes written as f16 {0,1}
// ---------------------------------------------------------------------------
__global__ void smlp_lif_kernel(const float* __restrict__ h,
                                _Float16* __restrict__ s, int plane, int T) {
  int p = blockIdx.x * blockDim.x + threadIdx.x;
  if (p >= plane) return;
  float v = 0.0f;
  for (int t = 0; t < T; ++t) {
    float x = h[(size_t)t * plane + p];
    v = 0.5f * (v + x);
    bool fire = (v >= 1.0f);
    s[(size_t)t * plane + p] = (_Float16)(fire ? 1.0f : 0.0f);
    v = fire ? 0.0f : v;
  }
}

// ---------------------------------------------------------------------------
// WMMA f16 GEMM with double-buffered async LDS staging:
//   C[M,N] = A[M,K] * W[N,K]^T
// Block = 256 threads = 8 waves (2x4); block tile 128x256; K-step 32.
// LDS rows padded to 40 halves (80B) -> fragment ds_load_b128 is 20-bank
// stride mod 64, conflict-free across the 16 lanes, still 16B-aligned.
// ---------------------------------------------------------------------------
#define BM  128
#define BN  256
#define BK  32
#define LDH 40  // padded LDS row stride in halves

__global__ __launch_bounds__(256) void smlp_gemm_wmma_f16(
    const _Float16* __restrict__ A, const _Float16* __restrict__ W,
    float* __restrict__ C, int M, int N, int K) {
  __shared__ __align__(16) _Float16 smem[2][(BM + BN) * LDH];

  const int tid  = threadIdx.x;
  const int lane = tid & 31;
  const int wave = tid >> 5;
  const int wm = wave & 1;   // 0..1 (M)
  const int wn = wave >> 1;  // 0..3 (N)
  const int mBlk = blockIdx.y * BM;
  const int nBlk = blockIdx.x * BN;
  const int hi = lane >> 4;
  const int l15 = lane & 15;

  v8f zero = {};
  v8f acc[4][4];
#pragma unroll
  for (int i = 0; i < 4; ++i)
#pragma unroll
    for (int j = 0; j < 4; ++j) acc[i][j] = zero;

  const int nK = K / BK;

  auto stageTile = [&](int buf, int k0) {
    _Float16* s = smem[buf];
    // A tile: 128 rows x 32 halves = 512 16B chunks, 2 per thread
#pragma unroll
    for (int it = 0; it < 2; ++it) {
      int c = tid + it * 256;
      int row = c >> 2, col = (c & 3) * 8;
      const _Float16* g = A + (size_t)(mBlk + row) * K + (k0 + col);
      __builtin_prefetch(g + 2 * BK, 0, 3);  // warm near caches, 2 K-steps out
      stage16(g, s + row * LDH + col);
    }
    // B tile: 256 rows x 32 halves = 1024 chunks, 4 per thread
#pragma unroll
    for (int it = 0; it < 4; ++it) {
      int c = tid + it * 256;
      int row = c >> 2, col = (c & 3) * 8;
      const _Float16* g = W + (size_t)(nBlk + row) * K + (k0 + col);
      __builtin_prefetch(g + 2 * BK, 0, 3);
      stage16(g, s + (BM + row) * LDH + col);
    }
  };

  int buf = 0;
  stageTile(0, 0);

  for (int kk = 0; kk < nK; ++kk) {
    wait_stage();      // own-wave asyncs done
    __syncthreads();   // all waves' staging visible; prior buffer reusable
    if (kk + 1 < nK) stageTile(buf ^ 1, (kk + 1) * BK);

    const _Float16* sA = smem[buf];
    const _Float16* sB = smem[buf] + BM * LDH;

    v16h af[4], bf[4];
#pragma unroll
    for (int i = 0; i < 4; ++i) {
      const _Float16* p = sA + (wm * 64 + i * 16 + l15) * LDH + hi * 8;
      union { v16h v; v8h h[2]; } u;
      u.h[0] = *(const v8h*)p;         // K = hi*8 + [0,8)
      u.h[1] = *(const v8h*)(p + 16);  // K = 16 + hi*8 + [0,8)
      af[i] = u.v;
    }
#pragma unroll
    for (int j = 0; j < 4; ++j) {
      const _Float16* p = sB + (wn * 64 + j * 16 + l15) * LDH + hi * 8;
      union { v16h v; v8h h[2]; } u;
      u.h[0] = *(const v8h*)p;
      u.h[1] = *(const v8h*)(p + 16);
      bf[j] = u.v;
    }
#pragma unroll
    for (int i = 0; i < 4; ++i)
#pragma unroll
      for (int j = 0; j < 4; ++j)
        acc[i][j] = __builtin_amdgcn_wmma_f32_16x16x32_f16(
            false, af[i], false, bf[j], (short)0, acc[i][j], false, false);

    buf ^= 1;
  }

  // store: VGPR r holds (M = r + 8*hi, N = l15) of each 16x16 tile
  const int mW = mBlk + wm * 64;
  const int nW = nBlk + wn * 64;
#pragma unroll
  for (int i = 0; i < 4; ++i) {
#pragma unroll
    for (int j = 0; j < 4; ++j) {
      const int col = nW + j * 16 + l15;
#pragma unroll
      for (int r = 0; r < 8; ++r) {
        const int row = mW + i * 16 + hi * 8 + r;
        C[(size_t)row * N + col] = acc[i][j][r];
      }
    }
  }
}

// ---------------------------------------------------------------------------
// Launch: x->f16, W*->f16, then GEMM1 -> LIF -> GEMM2 -> LIF -> GEMM3
// ---------------------------------------------------------------------------
extern "C" void kernel_launch(void* const* d_in, const int* in_sizes, int n_in,
                              void* d_out, int out_size, void* d_ws, size_t ws_size,
                              hipStream_t stream) {
  const float* x  = (const float*)d_in[0];  // [T,B,IN]
  const float* W1 = (const float*)d_in[1];  // [HID,IN]
  const float* W2 = (const float*)d_in[2];  // [HID,HID]
  const float* W3 = (const float*)d_in[3];  // [OUT,HID]

  const int T = 16, Bsz = 128, IN = 2048, HID = 4096, OUT = 1024;
  const int M = T * Bsz;  // 2048 rows, t-major

  char* ws = (char*)d_ws;
  _Float16* Xh  = (_Float16*)ws; ws += (size_t)M   * IN  * sizeof(_Float16);
  _Float16* W1h = (_Float16*)ws; ws += (size_t)HID * IN  * sizeof(_Float16);
  _Float16* W2h = (_Float16*)ws; ws += (size_t)HID * HID * sizeof(_Float16);
  _Float16* W3h = (_Float16*)ws; ws += (size_t)OUT * HID * sizeof(_Float16);
  float*    H   = (float*)ws;    ws += (size_t)M   * HID * sizeof(float);
  _Float16* S1  = (_Float16*)ws; ws += (size_t)M   * HID * sizeof(_Float16);
  _Float16* S2  = (_Float16*)ws;

  auto cvtBlocks = [](size_t n) { return (int)((n / 4 + 255) / 256); };
  smlp_cvt_f16_kernel<<<cvtBlocks((size_t)M * IN), 256, 0, stream>>>(x, Xh, M * IN);
  smlp_cvt_f16_kernel<<<cvtBlocks((size_t)HID * IN), 256, 0, stream>>>(W1, W1h, HID * IN);
  smlp_cvt_f16_kernel<<<cvtBlocks((size_t)HID * HID), 256, 0, stream>>>(W2, W2h, HID * HID);
  smlp_cvt_f16_kernel<<<cvtBlocks((size_t)OUT * HID), 256, 0, stream>>>(W3, W3h, OUT * HID);

  const dim3 blk(256);
  const int plane = Bsz * HID;
  const int lifBlocks = (plane + 255) / 256;

  smlp_gemm_wmma_f16<<<dim3(HID / BN, M / BM), blk, 0, stream>>>(
      Xh, W1h, H, M, HID, IN);
  smlp_lif_kernel<<<lifBlocks, 256, 0, stream>>>(H, S1, plane, T);

  smlp_gemm_wmma_f16<<<dim3(HID / BN, M / BM), blk, 0, stream>>>(
      S1, W2h, H, M, HID, HID);
  smlp_lif_kernel<<<lifBlocks, 256, 0, stream>>>(H, S2, plane, T);

  smlp_gemm_wmma_f16<<<dim3(OUT / BN, M / BM), blk, 0, stream>>>(
      S2, W3h, (float*)d_out, M, OUT, HID);
}
